// MultiHeadAttentionProj_40492951666706
// MI455X (gfx1250) — compile-verified, matrix operand order checked by reading, not run on hardware
//
#include <hip/hip_runtime.h>
#include <hip/hip_bf16.h>

// ---------------------------------------------------------------------------
// MI455X (gfx1250) multi-head attention, f16-WMMA everywhere, f32 accumulate.
// Pipeline: q,W->f16 | QKV proj (WMMA GEMM) | flash attention (WMMA) | out proj.
// ---------------------------------------------------------------------------

typedef __attribute__((ext_vector_type(16))) _Float16 v16h;
typedef __attribute__((ext_vector_type(8)))  _Float16 v8h;
typedef __attribute__((ext_vector_type(4)))  _Float16 v4h;
typedef __attribute__((ext_vector_type(8)))  float    v8f;
typedef __attribute__((ext_vector_type(4)))  float    v4f;

constexpr int Bc = 4, Nc = 2048, Dc = 512, Hc = 8, HDc = 64;
constexpr int BN = Bc * Nc;                 // 8192 rows

__device__ __forceinline__ v8f wmma_f16(v16h a, v16h b, v8f c) {
  // (neg_a, A, neg_b, B, c_mod, C, reuse_a, reuse_b)
  return __builtin_amdgcn_wmma_f32_16x16x32_f16(false, a, false, b, (short)0, c,
                                                false, false);
}

__device__ __forceinline__ v8f zero8() {
  v8f z = {0.f, 0.f, 0.f, 0.f, 0.f, 0.f, 0.f, 0.f};
  return z;
}

// --------------------------- f32 -> f16 convert ----------------------------
__global__ void cvt_f32_to_f16_kernel(const float* __restrict__ in,
                                      _Float16* __restrict__ out) {
  const size_t i = ((size_t)blockIdx.x * blockDim.x + threadIdx.x) * 4;
  v4f x = *(const v4f*)(in + i);
  v4h h;
  h[0] = (_Float16)x[0]; h[1] = (_Float16)x[1];
  h[2] = (_Float16)x[2]; h[3] = (_Float16)x[3];
  *(v4h*)(out + i) = h;
}

// ------------------------------ WMMA GEMM ----------------------------------
// C[BN, D] = X[BN, D](f16) @ W[D, D](f16, used as W^T).
// mode 0: write f16 head-major [H][B][N][HD] (for Q/K/V).
// mode 1: write f32 row-major  [BN][D]       (final output).
// One wave computes one 16x16 output tile; K loop in steps of 32.
// Inner loop is pure global_load_b128 -> v_wmma (weights pre-converted).
__global__ void gemm16_kernel(const _Float16* __restrict__ X,
                              const _Float16* __restrict__ W,
                              _Float16* __restrict__ outH,
                              float* __restrict__ outF,
                              int mode) {
  const int wave  = blockIdx.x * (blockDim.x >> 5) + (threadIdx.x >> 5);
  const int lane  = threadIdx.x & 31;
  const int tilesN = Dc / 16;                 // 32
  const int tm = wave / tilesN;
  const int tn = wave % tilesN;
  const int m0 = tm * 16, n0 = tn * 16;
  const int half = lane >> 4;                 // K-run selector (ISA 16-bit layout)
  const int idx  = lane & 15;                 // A: row, B: column

  const _Float16* ar = X + (size_t)(m0 + idx) * Dc;
  const _Float16* br = W + (size_t)(n0 + idx) * Dc;

  v8f acc = zero8();
  for (int k0 = 0; k0 < Dc; k0 += 32) {
    if (k0 + 64 < Dc) {                       // speculative WGP prefetch
      __builtin_prefetch(ar + k0 + 64, 0, 1);
      __builtin_prefetch(br + k0 + 64, 0, 1);
    }
    // A operand: row m=idx, runs K=[half*8..+8) and K=[16+half*8..+8)
    v8h a0 = *(const v8h*)(ar + k0 + half * 8);
    v8h a1 = *(const v8h*)(ar + k0 + 16 + half * 8);
    // B operand (= W^T tile): column n=idx, same K runs
    v8h b0 = *(const v8h*)(br + k0 + half * 8);
    v8h b1 = *(const v8h*)(br + k0 + 16 + half * 8);
    v16h A, Bm;
#pragma unroll
    for (int i = 0; i < 8; ++i) {
      A[i] = a0[i];  A[8 + i] = a1[i];
      Bm[i] = b0[i]; Bm[8 + i] = b1[i];
    }
    acc = wmma_f16(A, Bm, acc);
  }

  // C layout: lane holds column n=idx; VGPR r holds row m=r+8*half.
  if (mode == 0) {
#pragma unroll
    for (int r = 0; r < 8; ++r) {
      const int row  = m0 + r + 8 * half;      // index into [B*N]
      const int b    = row / Nc;
      const int nseq = row % Nc;
      const int d    = n0 + idx;
      const int h    = d / HDc;
      const int hd   = d % HDc;
      outH[((size_t)(h * Bc + b) * Nc + nseq) * HDc + hd] = (_Float16)acc[r];
    }
  } else {
#pragma unroll
    for (int r = 0; r < 8; ++r) {
      const int row = m0 + r + 8 * half;
      outF[(size_t)row * Dc + n0 + idx] = acc[r];
    }
  }
}

// ---------------------------- flash attention ------------------------------
// One wave per 16-query tile of one (h,b). 8 waves/block share (h,b), so the
// 32-key V tile is staged once per block into LDS (transposed [hd][key]).
__global__ void attn_kernel(const _Float16* __restrict__ Qh,   // [H][B][N][HD]
                            const _Float16* __restrict__ Kh,
                            const _Float16* __restrict__ Vh,
                            const float* __restrict__ bias,    // [B][N][N]
                            const int* __restrict__ mask,      // [B][N]
                            _Float16* __restrict__ Oh) {       // [B][N][D] f16
  __shared__ _Float16 VstT[HDc * 32];        // transposed V tile: [hd][keyLocal]
  __shared__ _Float16 Pst[8 * 16 * 32];      // per-wave P tile [16 q][32 key]

  const int waveId = threadIdx.x >> 5;
  const int lane   = threadIdx.x & 31;
  const int t  = blockIdx.x * 8 + waveId;    // 8 waves per block, same (h,b)
  const int qt = t % (Nc / 16);
  const int b  = (t / (Nc / 16)) % Bc;
  const int h  = t / (Bc * (Nc / 16));
  const int q0 = qt * 16;
  const int half = lane >> 4;
  const int idx  = lane & 15;

  const size_t hb = (size_t)(h * Bc + b) * Nc;
  const _Float16* Qb = Qh + hb * HDc;
  const _Float16* Kb = Kh + hb * HDc;
  const _Float16* Vb = Vh + hb * HDc;
  const float* biasQ = bias + ((size_t)b * Nc + q0) * Nc;
  const int*  maskB  = mask + b * Nc;
  _Float16* P = Pst + waveId * (16 * 32);

  // Q A-operands (K-dim = 64 -> two 16x32 chunks), pre-scaled by 1/sqrt(HD).
  v16h qa0, qa1;
  {
    const _Float16* qr = Qb + (size_t)(q0 + idx) * HDc;
    v8h r00 = *(const v8h*)(qr + half * 8);
    v8h r01 = *(const v8h*)(qr + 16 + half * 8);
    v8h r10 = *(const v8h*)(qr + 32 + half * 8);
    v8h r11 = *(const v8h*)(qr + 48 + half * 8);
    const _Float16 s = (_Float16)0.125f;     // 1/sqrt(64)
#pragma unroll
    for (int i = 0; i < 8; ++i) {
      qa0[i] = r00[i] * s; qa0[8 + i] = r01[i] * s;
      qa1[i] = r10[i] * s; qa1[8 + i] = r11[i] * s;
    }
  }

  v8f o0 = zero8(), o1 = zero8(), o2 = zero8(), o3 = zero8();
  float rmax[8], rsum[8];
#pragma unroll
  for (int r = 0; r < 8; ++r) { rmax[r] = -1e30f; rsum[r] = 0.f; }

  for (int kk = 0; kk < Nc; kk += 32) {
    if (kk + 32 < Nc) {                      // prefetch next key chunk (L2-hot)
      __builtin_prefetch(Kb + (size_t)(kk + 32 + idx) * HDc, 0, 1);
      __builtin_prefetch(Vb + (size_t)(kk + 32 + idx) * HDc, 0, 1);
    }
    __syncthreads();                         // prior tile fully consumed
    {                                        // stage V[32 keys][64 hd] transposed
      const int key = threadIdx.x >> 3;
      const int hd0 = (threadIdx.x & 7) * 8;
      v8h v = *(const v8h*)(Vb + (size_t)(kk + key) * HDc + hd0);
#pragma unroll
      for (int i = 0; i < 8; ++i) VstT[(hd0 + i) * 32 + key] = v[i];
    }
    __syncthreads();

    // scores: two 16x16 key tiles, each = 2 WMMAs over K-dim 64
    v8f s0 = zero8(), s1 = zero8();
    {
      const _Float16* kr = Kb + (size_t)(kk + idx) * HDc;       // tile 0 col=key
      v16h kb0, kb1;
      v8h c00 = *(const v8h*)(kr + half * 8);
      v8h c01 = *(const v8h*)(kr + 16 + half * 8);
      v8h c10 = *(const v8h*)(kr + 32 + half * 8);
      v8h c11 = *(const v8h*)(kr + 48 + half * 8);
#pragma unroll
      for (int i = 0; i < 8; ++i) {
        kb0[i] = c00[i]; kb0[8 + i] = c01[i];
        kb1[i] = c10[i]; kb1[8 + i] = c11[i];
      }
      s0 = wmma_f16(qa0, kb0, s0);
      s0 = wmma_f16(qa1, kb1, s0);
    }
    {
      const _Float16* kr = Kb + (size_t)(kk + 16 + idx) * HDc;  // tile 1
      v16h kb0, kb1;
      v8h c00 = *(const v8h*)(kr + half * 8);
      v8h c01 = *(const v8h*)(kr + 16 + half * 8);
      v8h c10 = *(const v8h*)(kr + 32 + half * 8);
      v8h c11 = *(const v8h*)(kr + 48 + half * 8);
#pragma unroll
      for (int i = 0; i < 8; ++i) {
        kb0[i] = c00[i]; kb0[8 + i] = c01[i];
        kb1[i] = c10[i]; kb1[8 + i] = c11[i];
      }
      s1 = wmma_f16(qa0, kb0, s1);
      s1 = wmma_f16(qa1, kb1, s1);
    }

    // bias + key mask (lane = column = key; rows r+8*half)
    const int mk0 = maskB[kk + idx];
    const int mk1 = maskB[kk + 16 + idx];
#pragma unroll
    for (int r = 0; r < 8; ++r) {
      const size_t qoff = (size_t)(r + 8 * half) * Nc;
      s0[r] = mk0 ? -1e30f : (s0[r] + biasQ[qoff + kk + idx]);
      s1[r] = mk1 ? -1e30f : (s1[r] + biasQ[qoff + kk + 16 + idx]);
    }

    // online softmax: row reductions across the 16-lane column groups
    float alpha[8];
#pragma unroll
    for (int r = 0; r < 8; ++r) {
      float m = fmaxf(s0[r], s1[r]);
#pragma unroll
      for (int sh = 8; sh >= 1; sh >>= 1) m = fmaxf(m, __shfl_xor(m, sh, 32));
      const float nm = fmaxf(rmax[r], m);
      const float al = __expf(rmax[r] - nm);
      const float p0 = __expf(s0[r] - nm);
      const float p1 = __expf(s1[r] - nm);
      float ps = p0 + p1;
#pragma unroll
      for (int sh = 8; sh >= 1; sh >>= 1) ps += __shfl_xor(ps, sh, 32);
      rsum[r] = rsum[r] * al + ps;
      rmax[r] = nm;
      alpha[r] = al;
      s0[r] = p0;
      s1[r] = p1;
    }
#pragma unroll
    for (int r = 0; r < 8; ++r) {
      o0[r] *= alpha[r]; o1[r] *= alpha[r];
      o2[r] *= alpha[r]; o3[r] *= alpha[r];
    }

    // transpose P (C layout -> A layout) through per-wave LDS
#pragma unroll
    for (int r = 0; r < 8; ++r) {
      P[(r + 8 * half) * 32 + idx]      = (_Float16)s0[r];
      P[(r + 8 * half) * 32 + 16 + idx] = (_Float16)s1[r];
    }
    v16h pa;
    {
      v8h p0 = *(const v8h*)(P + idx * 32 + half * 8);
      v8h p1 = *(const v8h*)(P + idx * 32 + 16 + half * 8);
#pragma unroll
      for (int i = 0; i < 8; ++i) { pa[i] = p0[i]; pa[8 + i] = p1[i]; }
    }

    // O[16x64] += P[16x32] @ V[32x64]: 4 column tiles
#pragma unroll
    for (int nt = 0; nt < 4; ++nt) {
      const int hd = nt * 16 + idx;          // B column = hd
      v8h u0 = *(const v8h*)(VstT + hd * 32 + half * 8);
      v8h u1 = *(const v8h*)(VstT + hd * 32 + 16 + half * 8);
      v16h bv;
#pragma unroll
      for (int i = 0; i < 8; ++i) { bv[i] = u0[i]; bv[8 + i] = u1[i]; }
      if (nt == 0) o0 = wmma_f16(pa, bv, o0);
      else if (nt == 1) o1 = wmma_f16(pa, bv, o1);
      else if (nt == 2) o2 = wmma_f16(pa, bv, o2);
      else o3 = wmma_f16(pa, bv, o3);
    }
  }

  // normalize and store in [B][N][D] f16 for the output projection
#pragma unroll
  for (int r = 0; r < 8; ++r) {
    const float inv = 1.0f / rsum[r];
    const int row = q0 + r + 8 * half;
    const size_t base = ((size_t)b * Nc + row) * Dc + h * HDc;
    Oh[base + 0 * 16 + idx] = (_Float16)(o0[r] * inv);
    Oh[base + 1 * 16 + idx] = (_Float16)(o1[r] * inv);
    Oh[base + 2 * 16 + idx] = (_Float16)(o2[r] * inv);
    Oh[base + 3 * 16 + idx] = (_Float16)(o3[r] * inv);
  }
}

// ------------------------------- launcher ----------------------------------
extern "C" void kernel_launch(void* const* d_in, const int* in_sizes, int n_in,
                              void* d_out, int out_size, void* d_ws,
                              size_t ws_size, hipStream_t stream) {
  const float* q    = (const float*)d_in[0];
  const int*   mask = (const int*)d_in[1];
  const float* bias = (const float*)d_in[2];
  const float* Wq   = (const float*)d_in[3];
  const float* Wk   = (const float*)d_in[4];
  const float* Wv   = (const float*)d_in[5];
  const float* Wo   = (const float*)d_in[6];
  float* out = (float*)d_out;

  const size_t elems = (size_t)BN * Dc;      // 4,194,304 per activation buffer
  const size_t welems = (size_t)Dc * Dc;     //   262,144 per weight matrix
  _Float16* Xh  = (_Float16*)d_ws;           // q in f16          (8 MB)
  _Float16* Qh  = Xh + elems;                // head-major Q      (8 MB)
  _Float16* Kh  = Qh + elems;                // head-major K      (8 MB)
  _Float16* Vh  = Kh + elems;                // head-major V      (8 MB)
  _Float16* Oh  = Vh + elems;                // attn out [B,N,D]  (8 MB)
  _Float16* Wqh = Oh + elems;                // f16 weights (0.5 MB each)
  _Float16* Wkh = Wqh + welems;
  _Float16* Wvh = Wkh + welems;
  _Float16* Woh = Wvh + welems;

  // 1) convert q and all weights to f16 (4 elements / thread)
  cvt_f32_to_f16_kernel<<<(int)(elems / (256 * 4)), 256, 0, stream>>>(q, Xh);
  const int wcvtBlocks = (int)(welems / (256 * 4));   // 256
  cvt_f32_to_f16_kernel<<<wcvtBlocks, 256, 0, stream>>>(Wq, Wqh);
  cvt_f32_to_f16_kernel<<<wcvtBlocks, 256, 0, stream>>>(Wk, Wkh);
  cvt_f32_to_f16_kernel<<<wcvtBlocks, 256, 0, stream>>>(Wv, Wvh);
  cvt_f32_to_f16_kernel<<<wcvtBlocks, 256, 0, stream>>>(Wo, Woh);

  // 2) QKV projections: out = X @ W^T, head-major f16
  const int gemmBlocks = (BN / 16) * (Dc / 16) / 8;   // 2048
  gemm16_kernel<<<gemmBlocks, 256, 0, stream>>>(Xh, Wqh, Qh, nullptr, 0);
  gemm16_kernel<<<gemmBlocks, 256, 0, stream>>>(Xh, Wkh, Kh, nullptr, 0);
  gemm16_kernel<<<gemmBlocks, 256, 0, stream>>>(Xh, Wvh, Vh, nullptr, 0);

  // 3) attention: 8 waves/block, block shares (h,b); 4096 waves total
  const int attnBlocks = Hc * Bc * (Nc / 16) / 8;     // 512
  attn_kernel<<<attnBlocks, 256, 0, stream>>>(Qh, Kh, Vh, bias, mask, Oh);

  // 4) output projection to f32
  gemm16_kernel<<<gemmBlocks, 256, 0, stream>>>(Oh, Woh, nullptr, out, 1);
}
